// GraphConvLayer_68822555951393
// MI455X (gfx1250) — matrix operand order repaired and checked by reading, not compiled, not run on hardware
//
#include <hip/hip_runtime.h>
#include <math.h>

#define N_NODES 50000
#define N_EDGES 800000
#define IN_DIM  128
#define OUT_DIM 128
#define EDGE_DIM 64
#define MSG_DIM 320   // 2*IN_DIM + EDGE_DIM
#define LN_EPS  1e-5f

typedef __attribute__((ext_vector_type(16))) _Float16 v16h;
typedef __attribute__((ext_vector_type(8)))  float    v8f;

// ---------- WMMA fragment loaders (two 16B loads each) ----------
// A 16x32 f16: lane half h: elems 0..7 <-> K = h*8..h*8+7 ; elems 8..15 <-> K = 16+h*8..
__device__ __forceinline__ v16h load_fragA(const _Float16* row, int half_) {
  v16h r;
  ((uint4*)&r)[0] = *(const uint4*)(row + half_ * 8);
  ((uint4*)&r)[1] = *(const uint4*)(row + 16 + half_ * 8);
  return r;
}
// B 32x16 f16: lane = N, elems 0..7 <-> K = h*8..h*8+7 ; elems 8..15 <-> K = 16+h*8..
__device__ __forceinline__ v16h load_fragB(const _Float16* row, int half_) {
  v16h r;
  ((uint4*)&r)[0] = *(const uint4*)(row + half_ * 8);
  ((uint4*)&r)[1] = *(const uint4*)(row + 16 + half_ * 8);
  return r;
}

union H4 { _Float16 h[4]; uint2 u; };

// ---------- weight prep: transpose + f32->f16 ----------
__global__ void prep_weights(const float* __restrict__ W1, const float* __restrict__ W2,
                             _Float16* __restrict__ W1t, _Float16* __restrict__ W2t) {
  int i = blockIdx.x * blockDim.x + threadIdx.x;
  if (i < MSG_DIM * OUT_DIM) {                       // W1 (320x128) -> W1t (128x320)
    int k = i / OUT_DIM, n = i % OUT_DIM;
    W1t[n * MSG_DIM + k] = (_Float16)W1[i];
  }
  int j = i - MSG_DIM * OUT_DIM;
  if (j >= 0 && j < OUT_DIM * OUT_DIM) {             // W2 (128x128) -> W2t (128x128)
    int k = j / OUT_DIM, n = j % OUT_DIM;
    W2t[n * OUT_DIM + k] = (_Float16)W2[j];
  }
}

// ---------- stage 16-edge tile into LDS (f16) + GEMM1 (ReLU) into hT ----------
__device__ __forceinline__ void stage_and_gemm1(
    const float* __restrict__ x, const int* __restrict__ ei,
    const float* __restrict__ ea, const _Float16* __restrict__ W1t,
    const float* __restrict__ b1,
    _Float16* msg, _Float16* hT, int ebase, int lane) {
  const int half_ = lane >> 4, ln = lane & 15;
  #pragma unroll 4
  for (int e = 0; e < 16; ++e) {
    int sidx = ei[ebase + e];
    int didx = ei[N_EDGES + ebase + e];
    float4 xd = *(const float4*)(x + (long)didx * IN_DIM + lane * 4);
    float4 xs = *(const float4*)(x + (long)sidx * IN_DIM + lane * 4);
    _Float16* mrow = msg + e * MSG_DIM;
    H4 t0; t0.h[0] = (_Float16)xd.x; t0.h[1] = (_Float16)xd.y;
           t0.h[2] = (_Float16)xd.z; t0.h[3] = (_Float16)xd.w;
    *(uint2*)(mrow + lane * 4) = t0.u;
    H4 t1; t1.h[0] = (_Float16)(xs.x - xd.x); t1.h[1] = (_Float16)(xs.y - xd.y);
           t1.h[2] = (_Float16)(xs.z - xd.z); t1.h[3] = (_Float16)(xs.w - xd.w);
    *(uint2*)(mrow + IN_DIM + lane * 4) = t1.u;
    if (lane < 16) {
      float4 av = *(const float4*)(ea + (long)(ebase + e) * EDGE_DIM + lane * 4);
      H4 t2; t2.h[0] = (_Float16)av.x; t2.h[1] = (_Float16)av.y;
             t2.h[2] = (_Float16)av.z; t2.h[3] = (_Float16)av.w;
      *(uint2*)(mrow + 2 * IN_DIM + lane * 4) = t2.u;
    }
  }
  asm volatile("s_wait_dscnt 0" ::: "memory");
  // GEMM1: (16x320) @ (320x128), ReLU, f16 result into hT
  for (int nt = 0; nt < 8; ++nt) {
    v8f c = {};
    const _Float16* arow = msg + ln * MSG_DIM;
    const _Float16* brow = W1t + (nt * 16 + ln) * MSG_DIM;
    #pragma unroll
    for (int ks = 0; ks < MSG_DIM / 32; ++ks) {
      v16h a = load_fragA(arow + ks * 32, half_);
      v16h b = load_fragB(brow + ks * 32, half_);
      c = __builtin_amdgcn_wmma_f32_16x16x32_f16(false, a, false, b, (short)0, c, false, false);
    }
    int n = nt * 16 + ln;
    float bias = b1[n];
    #pragma unroll
    for (int v = 0; v < 8; ++v) {
      float f = c[v] + bias;
      f = f > 0.f ? f : 0.f;
      hT[(v + 8 * half_) * OUT_DIM + n] = (_Float16)f;   // D frag: M = v + 8*half, N = n
    }
  }
  asm volatile("s_wait_dscnt 0" ::: "memory");
}

// ---------- pass 1: messages -> attention scores ----------
__global__ void __launch_bounds__(128) msg_score_kernel(
    const float* __restrict__ x, const int* __restrict__ ei, const float* __restrict__ ea,
    const _Float16* __restrict__ W1t, const _Float16* __restrict__ W2t,
    const float* __restrict__ b1, const float* __restrict__ b2,
    const float* __restrict__ Wa, const float* __restrict__ ba,
    float* __restrict__ scores) {
  __shared__ _Float16 sMsg[4][16 * MSG_DIM];
  __shared__ _Float16 sH[4][16 * OUT_DIM];
  const int wave = threadIdx.x >> 5, lane = threadIdx.x & 31;
  const int half_ = lane >> 4, ln = lane & 15;
  const int ebase = (blockIdx.x * 4 + wave) * 16;
  stage_and_gemm1(x, ei, ea, W1t, b1, sMsg[wave], sH[wave], ebase, lane);

  float pm[8] = {0.f, 0.f, 0.f, 0.f, 0.f, 0.f, 0.f, 0.f};
  for (int nt = 0; nt < 8; ++nt) {
    v8f c = {};
    const _Float16* arow = sH[wave] + ln * OUT_DIM;
    const _Float16* brow = W2t + (nt * 16 + ln) * OUT_DIM;
    #pragma unroll
    for (int ks = 0; ks < 4; ++ks) {
      v16h a = load_fragA(arow + ks * 32, half_);
      v16h b = load_fragB(brow + ks * 32, half_);
      c = __builtin_amdgcn_wmma_f32_16x16x32_f16(false, a, false, b, (short)0, c, false, false);
    }
    int n = nt * 16 + ln;
    float wa = Wa[n], bb = b2[n];
    #pragma unroll
    for (int v = 0; v < 8; ++v) pm[v] += (c[v] + bb) * wa;   // partial dot(messages, Wa)
  }
  // reduce each pm[v] across the 16 lanes of this half (same M row)
  float myscore = 0.f;
  #pragma unroll
  for (int v = 0; v < 8; ++v) {
    float s = pm[v];
    s += __shfl_xor(s, 1); s += __shfl_xor(s, 2);
    s += __shfl_xor(s, 4); s += __shfl_xor(s, 8);
    if (ln == v) myscore = s;
  }
  if (ln < 8) {
    int m = ln + 8 * half_;
    float s = myscore + ba[0];
    scores[ebase + m] = s > 0.f ? s : 0.2f * s;              // LeakyReLU(0.2)
  }
}

// ---------- global softmax over all edges ----------
__device__ __forceinline__ unsigned enc_f(float f) {
  unsigned u = __float_as_uint(f);
  return (u >> 31) ? ~u : (u | 0x80000000u);
}
__device__ __forceinline__ float dec_f(unsigned u) {
  return (u >> 31) ? __uint_as_float(u ^ 0x80000000u) : __uint_as_float(~u);
}

__global__ void init_scalars(unsigned* gmax, float* gsum) { *gmax = 0u; *gsum = 0.f; }

__global__ void __launch_bounds__(256) reduce_max_kernel(const float* __restrict__ s,
                                                         unsigned* __restrict__ gmax) {
  __shared__ unsigned sm[256];
  unsigned best = 0u;
  for (int i = blockIdx.x * blockDim.x + threadIdx.x; i < N_EDGES; i += gridDim.x * blockDim.x) {
    unsigned e = enc_f(s[i]);
    best = e > best ? e : best;
  }
  sm[threadIdx.x] = best; __syncthreads();
  for (int st = 128; st > 0; st >>= 1) {
    if (threadIdx.x < st) { unsigned o = sm[threadIdx.x + st]; if (o > sm[threadIdx.x]) sm[threadIdx.x] = o; }
    __syncthreads();
  }
  if (threadIdx.x == 0) atomicMax(gmax, sm[0]);
}

__global__ void __launch_bounds__(256) reduce_sum_kernel(const float* __restrict__ s,
                                                         const unsigned* __restrict__ gmax,
                                                         float* __restrict__ gsum) {
  __shared__ float sm[256];
  float mx = dec_f(*gmax);
  float acc = 0.f;
  for (int i = blockIdx.x * blockDim.x + threadIdx.x; i < N_EDGES; i += gridDim.x * blockDim.x)
    acc += __expf(s[i] - mx);
  sm[threadIdx.x] = acc; __syncthreads();
  for (int st = 128; st > 0; st >>= 1) {
    if (threadIdx.x < st) sm[threadIdx.x] += sm[threadIdx.x + st];
    __syncthreads();
  }
  if (threadIdx.x == 0) unsafeAtomicAdd(gsum, sm[0]);
}

__global__ void __launch_bounds__(256) normalize_kernel(float* __restrict__ s,
                                                        const unsigned* __restrict__ gmax,
                                                        const float* __restrict__ gsum) {
  int i = blockIdx.x * blockDim.x + threadIdx.x;
  float mx = dec_f(*gmax);
  float inv = 1.f / *gsum;
  if (i < N_EDGES) s[i] = __expf(s[i] - mx) * inv;
}

// ---------- pass 2: recompute messages, weight, scatter-add into agg(=d_out) ----------
__global__ void __launch_bounds__(128) msg_scatter_kernel(
    const float* __restrict__ x, const int* __restrict__ ei, const float* __restrict__ ea,
    const _Float16* __restrict__ W1t, const _Float16* __restrict__ W2t,
    const float* __restrict__ b1, const float* __restrict__ b2,
    const float* __restrict__ weights, float* __restrict__ agg) {
  __shared__ _Float16 sMsg[4][16 * MSG_DIM];
  __shared__ _Float16 sH[4][16 * OUT_DIM];
  const int wave = threadIdx.x >> 5, lane = threadIdx.x & 31;
  const int half_ = lane >> 4, ln = lane & 15;
  const int ebase = (blockIdx.x * 4 + wave) * 16;

  float wv = 0.f; int dv = 0;
  if (lane < 16) { wv = weights[ebase + lane]; dv = ei[N_EDGES + ebase + lane]; }

  stage_and_gemm1(x, ei, ea, W1t, b1, sMsg[wave], sH[wave], ebase, lane);

  for (int nt = 0; nt < 8; ++nt) {
    v8f c = {};
    const _Float16* arow = sH[wave] + ln * OUT_DIM;
    const _Float16* brow = W2t + (nt * 16 + ln) * OUT_DIM;
    #pragma unroll
    for (int ks = 0; ks < 4; ++ks) {
      v16h a = load_fragA(arow + ks * 32, half_);
      v16h b = load_fragB(brow + ks * 32, half_);
      c = __builtin_amdgcn_wmma_f32_16x16x32_f16(false, a, false, b, (short)0, c, false, false);
    }
    int n = nt * 16 + ln;
    float bb = b2[n];
    #pragma unroll
    for (int v = 0; v < 8; ++v) {
      int m = v + 8 * half_;
      float w = __shfl(wv, m);
      int di = __shfl(dv, m);
      unsafeAtomicAdd(agg + (long)di * OUT_DIM + n, (c[v] + bb) * w);
    }
  }
}

// ---------- residual + LayerNorm (in-place on d_out which holds agg) ----------
__global__ void __launch_bounds__(256) ln_kernel(const float* __restrict__ x,
                                                 const float* __restrict__ gamma,
                                                 const float* __restrict__ beta,
                                                 float* __restrict__ out) {
  const int wave = threadIdx.x >> 5, lane = threadIdx.x & 31;
  const int node = blockIdx.x * 8 + wave;
  const long off = (long)node * OUT_DIM + lane * 4;
  float4 xv = *(const float4*)(x + off);
  float4 av = *(const float4*)(out + off);
  float o0 = xv.x + av.x, o1 = xv.y + av.y, o2 = xv.z + av.z, o3 = xv.w + av.w;
  float s  = o0 + o1 + o2 + o3;
  float ss = o0 * o0 + o1 * o1 + o2 * o2 + o3 * o3;
  #pragma unroll
  for (int m = 1; m < 32; m <<= 1) { s += __shfl_xor(s, m); ss += __shfl_xor(ss, m); }
  float mu  = s * (1.f / 128.f);
  float var = ss * (1.f / 128.f) - mu * mu;
  float inv = rsqrtf(var + LN_EPS);
  float4 g  = *(const float4*)(gamma + lane * 4);
  float4 bt = *(const float4*)(beta + lane * 4);
  float4 r;
  r.x = (o0 - mu) * inv * g.x + bt.x;
  r.y = (o1 - mu) * inv * g.y + bt.y;
  r.z = (o2 - mu) * inv * g.z + bt.z;
  r.w = (o3 - mu) * inv * g.w + bt.w;
  *(float4*)(out + off) = r;
}

extern "C" void kernel_launch(void* const* d_in, const int* in_sizes, int n_in,
                              void* d_out, int out_size, void* d_ws, size_t ws_size,
                              hipStream_t stream) {
  const float* x     = (const float*)d_in[0];
  const int*   ei    = (const int*)d_in[1];
  const float* ea    = (const float*)d_in[2];
  const float* W1    = (const float*)d_in[3];
  const float* b1    = (const float*)d_in[4];
  const float* W2    = (const float*)d_in[5];
  const float* b2    = (const float*)d_in[6];
  const float* Wa    = (const float*)d_in[7];
  const float* ba    = (const float*)d_in[8];
  const float* gamma = (const float*)d_in[9];
  const float* beta  = (const float*)d_in[10];
  float* out = (float*)d_out;

  char* ws = (char*)d_ws;
  float* scores = (float*)ws;                                  // E floats (scores -> weights)
  size_t off = ((size_t)N_EDGES * sizeof(float) + 255) & ~(size_t)255;
  unsigned* gmax = (unsigned*)(ws + off); off += 256;
  float*    gsum = (float*)(ws + off);    off += 256;
  _Float16* W1t  = (_Float16*)(ws + off); off += (size_t)MSG_DIM * OUT_DIM * 2;
  off = (off + 255) & ~(size_t)255;
  _Float16* W2t  = (_Float16*)(ws + off); off += (size_t)OUT_DIM * OUT_DIM * 2;

  prep_weights<<<(MSG_DIM * OUT_DIM + OUT_DIM * OUT_DIM + 255) / 256, 256, 0, stream>>>(W1, W2, W1t, W2t);
  init_scalars<<<1, 1, 0, stream>>>(gmax, gsum);
  msg_score_kernel<<<N_EDGES / 64, 128, 0, stream>>>(x, ei, ea, W1t, W2t, b1, b2, Wa, ba, scores);
  reduce_max_kernel<<<512, 256, 0, stream>>>(scores, gmax);
  reduce_sum_kernel<<<512, 256, 0, stream>>>(scores, gmax, gsum);
  normalize_kernel<<<N_EDGES / 256, 256, 0, stream>>>(scores, gmax, gsum);
  hipMemsetAsync(d_out, 0, (size_t)N_NODES * OUT_DIM * sizeof(float), stream);
  msg_scatter_kernel<<<N_EDGES / 64, 128, 0, stream>>>(x, ei, ea, W1t, W2t, b1, b2, scores, out);
  ln_kernel<<<N_NODES / 8, 256, 0, stream>>>(x, gamma, beta, out);
}
